// MultiHeadAttention_31421980737929
// MI455X (gfx1250) — matrix operand order repaired and checked by reading, not compiled, not run on hardware
//
#include <hip/hip_runtime.h>
#include <math.h>

#define BATCH 2
#define SEQ   2048
#define HID   1024
#define NH    16
#define HD    64

typedef __attribute__((ext_vector_type(16))) _Float16 v16h;
typedef __attribute__((ext_vector_type(8)))  _Float16 v8h;
typedef __attribute__((ext_vector_type(8)))  float    v8f;

// ---------------------------------------------------------------------------
// WMMA fragment helpers (wave32, 16x16x32 f16 -> f32)
// A (16xK=32, 16-bit) per ISA layout: lane r=lane&15 holds row M=r,
//   lane-half hi=lane>>4 selects K-offset: elems 0..7 = K hi*8..hi*8+7,
//   elems 8..15 = K hi*8+16..hi*8+23.
// ---------------------------------------------------------------------------
__device__ __forceinline__ v16h load_fragA(const _Float16* base, int ld, int lane) {
    const int r  = lane & 15;
    const int hi = lane >> 4;
    const _Float16* p = base + r * ld + hi * 8;
    v8h lo = *(const v8h*)(p);
    v8h hh = *(const v8h*)(p + 16);
    v16h out;
#pragma unroll
    for (int i = 0; i < 8; ++i) { out[i] = lo[i]; out[i + 8] = hh[i]; }
    return out;
}

// B (K=32 x 16, 16-bit): lane n=lane&15 holds column N=n; lane-half hi selects
// K-offset: elems 0..7 = K hi*8..hi*8+7, elems 8..15 = K hi*8+16..hi*8+23.
// `base` points at a (N,K) row-major matrix tile (i.e. B^T), so each lane reads
// two 16-byte chunks.
__device__ __forceinline__ v16h load_fragB(const _Float16* base, int ld, int lane) {
    const int n  = lane & 15;
    const int hi = lane >> 4;
    const _Float16* p = base + n * ld + hi * 8;
    v8h lo = *(const v8h*)(p);
    v8h hh = *(const v8h*)(p + 16);
    v16h out;
#pragma unroll
    for (int i = 0; i < 8; ++i) { out[i] = lo[i]; out[i + 8] = hh[i]; }
    return out;
}

__device__ __forceinline__ v8f wmma_f16f32(v16h a, v16h b, v8f c) {
    return __builtin_amdgcn_wmma_f32_16x16x32_f16(
        /*neg_a=*/false, a, /*neg_b=*/false, b,
        /*c_mod=*/(short)0, c, /*reuse_a=*/false, /*reuse_b=*/false);
}

// ---------------------------------------------------------------------------
// Kernel 0: fp32 -> f16 convert
// ---------------------------------------------------------------------------
__global__ void mha_cvt_f16(const float* __restrict__ in, _Float16* __restrict__ out, int n) {
    int i = blockIdx.x * blockDim.x + threadIdx.x;
    if (i < n) out[i] = (_Float16)in[i];
}

// ---------------------------------------------------------------------------
// Kernel 1/4: C(M,N) f32 = A(M,K) f16 row-major  x  Bt(N,K) f16 row-major
// Register-blocked: one wave computes a 64x64 C tile (16 accumulators,
// 4 A-fragments x 4 B-fragments -> 16 WMMAs per k-step). Block = 4 waves
// stacked in M. grid = (N/64, M/256), block = 128.
// ---------------------------------------------------------------------------
__global__ __launch_bounds__(128)
void mha_gemm_f16(const _Float16* __restrict__ A, const _Float16* __restrict__ Bt,
                  float* __restrict__ C, int M, int N, int K) {
    const int lane = threadIdx.x & 31;
    const int wave = threadIdx.x >> 5;
    const int m0 = (blockIdx.y * 4 + wave) * 64;
    const int n0 = blockIdx.x * 64;
    if (m0 >= M) return;  // wave-uniform

    v8f acc[4][4] = {};
    for (int k0 = 0; k0 < K; k0 += 32) {
        v16h a[4], b[4];
#pragma unroll
        for (int i = 0; i < 4; ++i) {
            a[i] = load_fragA(A + (size_t)(m0 + i * 16) * K + k0, K, lane);
            b[i] = load_fragB(Bt + (size_t)(n0 + i * 16) * K + k0, K, lane);
        }
#pragma unroll
        for (int mi = 0; mi < 4; ++mi)
#pragma unroll
            for (int ni = 0; ni < 4; ++ni)
                acc[mi][ni] = wmma_f16f32(a[mi], b[ni], acc[mi][ni]);
    }
    const int rbase = (lane >> 4) << 3;
    const int cbase = lane & 15;
#pragma unroll
    for (int mi = 0; mi < 4; ++mi)
#pragma unroll
        for (int ni = 0; ni < 4; ++ni)
#pragma unroll
            for (int r = 0; r < 8; ++r)
                C[(size_t)(m0 + mi * 16 + rbase + r) * N + n0 + ni * 16 + cbase] =
                    acc[mi][ni][r];
}

// ---------------------------------------------------------------------------
// Kernel 2: RoPE + head split. qkv32 is (B*S, 3*HID) fp32.
// Outputs (f16): qh,kh as [b][h][s][d] (row-major per head),
//                vT as [b][h][d][s] so AV B-fragments are K-contiguous.
// One thread per (b,h,s).
// ---------------------------------------------------------------------------
__global__ void mha_rope_split(const float* __restrict__ qkv,
                               _Float16* __restrict__ qh,
                               _Float16* __restrict__ kh,
                               _Float16* __restrict__ vT) {
    const int idx = blockIdx.x * blockDim.x + threadIdx.x;   // (b*NH+h)*SEQ + s
    const int s = idx & (SEQ - 1);
    const int h = (idx >> 11) & (NH - 1);
    const int b = idx >> 15;
    const int bh = b * NH + h;

    const float* row = qkv + (size_t)(b * SEQ + s) * (3 * HID);
    const float* qp = row + h * HD;
    const float* kp = row + HID + h * HD;
    const float* vp = row + 2 * HID + h * HD;
    _Float16* qo = qh + (size_t)idx * HD;
    _Float16* ko = kh + (size_t)idx * HD;
    _Float16* vo = vT + (size_t)bh * HD * SEQ + s;

    const float C0 = -0.28782313662425572f;  // -2*ln(10000)/HD
    const float sf = (float)s;
#pragma unroll
    for (int j = 0; j < 16; ++j) {
        float f1 = sf * __expf((float)j * C0);
        float f2 = sf * __expf((float)(j + 16) * C0);
        float c1 = __cosf(f1), s1 = __sinf(f1);
        float c2 = __cosf(f2), s2 = __sinf(f2);
        float qa = qp[j], qb = qp[j + 16];
        float ka = kp[j], kb = kp[j + 16];
        qo[j]      = (_Float16)(qa * c1 - qb * s1);
        qo[j + 16] = (_Float16)(qb * c2 + qa * s2);
        ko[j]      = (_Float16)(ka * c1 - kb * s1);
        ko[j + 16] = (_Float16)(kb * c2 + ka * s2);
    }
#pragma unroll
    for (int d = 32; d < HD; ++d) { qo[d] = (_Float16)qp[d]; ko[d] = (_Float16)kp[d]; }
#pragma unroll
    for (int d = 0; d < HD; ++d) vo[(size_t)d * SEQ] = (_Float16)vp[d];
}

// ---------------------------------------------------------------------------
// Kernel 3: attention for one (b, h, 16-query block). 4 waves / 128 threads.
// LDS: 16x2048 f16 score/prob strip (64KB) + reduction scratch.
// ---------------------------------------------------------------------------
#define ATTN_SMEM (16 * SEQ * 2 + 128 * 4 + 16 * 4 + 16 * 4)

__global__ __launch_bounds__(128)
void mha_attn(const _Float16* __restrict__ qh,
              const _Float16* __restrict__ kh,
              const _Float16* __restrict__ vT,
              _Float16* __restrict__ ctx) {
    extern __shared__ char smem[];
    _Float16* Sb   = (_Float16*)smem;                 // [16][SEQ]
    float* red     = (float*)(smem + 16 * SEQ * 2);   // [16][8]
    float* rowmax  = red + 128;                       // [16]
    float* rowsum  = rowmax + 16;                     // [16]

    const int lane = threadIdx.x & 31;
    const int wave = threadIdx.x >> 5;
    const int q0 = blockIdx.x * 16;
    const int h  = blockIdx.y;
    const int b  = blockIdx.z;
    const int bh = b * NH + h;

    const _Float16* Qbase = qh + ((size_t)bh * SEQ + q0) * HD;
    const _Float16* Kbase = kh + (size_t)bh * SEQ * HD;
    const _Float16* Vbase = vT + (size_t)bh * HD * SEQ;

    const int r0 = (lane >> 4) << 3;   // C-tile row base for this lane-half
    const int cc = lane & 15;          // C-tile column for this lane

    // ---- Phase 1: S = (Q K^T) * 1/sqrt(hd), wave w covers keys [w*512, w*512+512)
    v16h qa0 = load_fragA(Qbase + 0,  HD, lane);
    v16h qa1 = load_fragA(Qbase + 32, HD, lane);
    for (int kt = wave * 32; kt < wave * 32 + 32; ++kt) {
        const int kk0 = kt * 16;
        v8f acc = {};
        v16h b0 = load_fragB(Kbase + (size_t)kk0 * HD + 0,  HD, lane);
        acc = wmma_f16f32(qa0, b0, acc);
        v16h b1 = load_fragB(Kbase + (size_t)kk0 * HD + 32, HD, lane);
        acc = wmma_f16f32(qa1, b1, acc);
#pragma unroll
        for (int r = 0; r < 8; ++r)
            Sb[(r0 + r) * SEQ + kk0 + cc] = (_Float16)(acc[r] * 0.125f);
    }
    __syncthreads();

    // ---- Phase 2: row softmax (8 threads per row, 256 cols each); in-place exp
    {
        const int row = threadIdx.x >> 3;
        const int sub = threadIdx.x & 7;
        _Float16* srow = Sb + row * SEQ + sub * 256;
        float m = -1e30f;
        for (int i = 0; i < 256; ++i) m = fmaxf(m, (float)srow[i]);
        red[row * 8 + sub] = m;
        __syncthreads();
        if (sub == 0) {
            float mm = red[row * 8];
            for (int i = 1; i < 8; ++i) mm = fmaxf(mm, red[row * 8 + i]);
            rowmax[row] = mm;
        }
        __syncthreads();
        m = rowmax[row];
        float ssum = 0.f;
        for (int i = 0; i < 256; ++i) {
            float e = __expf((float)srow[i] - m);
            srow[i] = (_Float16)e;        // unnormalized prob, f16
            ssum += e;
        }
        red[row * 8 + sub] = ssum;
        __syncthreads();
        if (sub == 0) {
            float ss = 0.f;
            for (int i = 0; i < 8; ++i) ss += red[row * 8 + i];
            rowsum[row] = ss;
        }
    }
    __syncthreads();

    // ---- Phase 3: O = P V ; wave w owns head-dim tile d0 = w*16
    {
        const int d0 = wave * 16;
        v8f acc = {};
        for (int k0 = 0; k0 < SEQ; k0 += 32) {
            v16h a = load_fragA(Sb + k0, SEQ, lane);                        // from LDS
            v16h bb = load_fragB(Vbase + (size_t)d0 * SEQ + k0, SEQ, lane); // V^T rows
            acc = wmma_f16f32(a, bb, acc);
        }
        _Float16* out = ctx + ((size_t)(b * SEQ + q0)) * HID + h * HD + d0;
#pragma unroll
        for (int r = 0; r < 8; ++r) {
            const int row = r0 + r;
            float v = acc[r] / rowsum[row];
            out[(size_t)row * HID + cc] = (_Float16)v;
        }
    }
}

// ---------------------------------------------------------------------------
// Host launcher
// ---------------------------------------------------------------------------
extern "C" void kernel_launch(void* const* d_in, const int* in_sizes, int n_in,
                              void* d_out, int out_size, void* d_ws, size_t ws_size,
                              hipStream_t stream) {
    (void)in_sizes; (void)n_in; (void)out_size; (void)ws_size;
    const float* x     = (const float*)d_in[0];   // (B,S,H)
    const float* Wqkv  = (const float*)d_in[1];   // (3H,H)
    const float* Wproj = (const float*)d_in[2];   // (H,H)
    float* out = (float*)d_out;                   // (B,S,H)

    char* ws = (char*)d_ws;
    _Float16* xh     = (_Float16*)ws; ws += (size_t)BATCH * SEQ * HID * 2;
    _Float16* wqkvh  = (_Float16*)ws; ws += (size_t)3 * HID * HID * 2;
    _Float16* wprojh = (_Float16*)ws; ws += (size_t)HID * HID * 2;
    float*    qkv32  = (float*)ws;    ws += (size_t)BATCH * SEQ * 3 * HID * 4;
    _Float16* q_h    = (_Float16*)ws; ws += (size_t)BATCH * NH * SEQ * HD * 2;
    _Float16* k_h    = (_Float16*)ws; ws += (size_t)BATCH * NH * SEQ * HD * 2;
    _Float16* vT_h   = (_Float16*)ws; ws += (size_t)BATCH * NH * HD * SEQ * 2;
    _Float16* ctx_h  = (_Float16*)ws; ws += (size_t)BATCH * SEQ * HID * 2;

    const int nX = BATCH * SEQ * HID;
    const int nWq = 3 * HID * HID;
    const int nWp = HID * HID;
    mha_cvt_f16<<<(nX + 255) / 256, 256, 0, stream>>>(x, xh, nX);
    mha_cvt_f16<<<(nWq + 255) / 256, 256, 0, stream>>>(Wqkv, wqkvh, nWq);
    mha_cvt_f16<<<(nWp + 255) / 256, 256, 0, stream>>>(Wproj, wprojh, nWp);

    // QKV projection: (4096 x 3072 x 1024), 64x64 tile per wave
    mha_gemm_f16<<<dim3(3 * HID / 64, (BATCH * SEQ) / 256), 128, 0, stream>>>(
        xh, wqkvh, qkv32, BATCH * SEQ, 3 * HID, HID);

    // RoPE + head split
    mha_rope_split<<<(BATCH * NH * SEQ) / 256, 256, 0, stream>>>(qkv32, q_h, k_h, vT_h);

    // Attention
    (void)hipFuncSetAttribute((const void*)mha_attn,
                              hipFuncAttributeMaxDynamicSharedMemorySize, ATTN_SMEM);
    mha_attn<<<dim3(SEQ / 16, NH, BATCH), 128, ATTN_SMEM, stream>>>(q_h, k_h, vT_h, ctx_h);

    // Output projection: (4096 x 1024 x 1024) -> fp32 d_out
    mha_gemm_f16<<<dim3(HID / 64, (BATCH * SEQ) / 256), 128, 0, stream>>>(
        ctx_h, wprojh, out, BATCH * SEQ, HID, HID);
}
